// FusedMoETensorCast_24352464569736
// MI455X (gfx1250) — matrix operand order repaired and checked by reading, not compiled, not run on hardware
//
#include <hip/hip_runtime.h>
#include <stdint.h>

// ---------------------------------------------------------------------------
// Fused MoE (gated-SiLU experts), MI455X / gfx1250, bf16 WMMA f32-accumulate.
// E=16 experts, K=2 topk, D=1024, F=512, T=4096, C=2048 capacity.
// ---------------------------------------------------------------------------

#define E_ 16
#define K_ 2
#define D_ 1024
#define F_ 512
#define T_ 4096
#define C_ 2048

#define LDA 40   // LDS row stride in bf16 (32 data + 8 pad = 80B, 16B aligned)

typedef __bf16 bf16;
typedef __attribute__((ext_vector_type(16))) __bf16 v16bf;
typedef __attribute__((ext_vector_type(8)))  float  v8f;
typedef __attribute__((ext_vector_type(4)))  int    i4;

union Frag16 { v16bf v; uint4 q[2]; };
union Pack16 { bf16 h[16]; uint4 q[2]; };
union Pack2  { bf16 h[2];  unsigned u; };

// ---- optional CDNA5 async global->LDS copy path ---------------------------
#if defined(__has_builtin)
#if __has_builtin(__builtin_amdgcn_global_load_async_to_lds_b128)
#define USE_ASYNC_LDS_COPY 1
#endif
#endif

#ifdef USE_ASYNC_LDS_COPY
typedef __attribute__((address_space(1))) i4* gp_i4;   // global int4*
typedef __attribute__((address_space(3))) i4* lp_i4;   // LDS int4*
#endif

__device__ __forceinline__ void wait_async0() {
#if defined(__has_builtin) && __has_builtin(__builtin_amdgcn_s_wait_asynccnt)
    __builtin_amdgcn_s_wait_asynccnt(0);
#else
    asm volatile("s_wait_asynccnt 0x0" ::: "memory");
#endif
}

// ---------------------------------------------------------------------------
// Kernel 0: zero output, reset routing scratch (ws is poisoned by harness).
// ---------------------------------------------------------------------------
__global__ __launch_bounds__(256) void moe_init(float* __restrict__ out,
                                                int* __restrict__ row_token,
                                                int* __restrict__ counts) {
    const int i = blockIdx.x * blockDim.x + threadIdx.x;
    if (i < T_ * D_) out[i] = 0.0f;
    if (i < E_ * C_) row_token[i] = -1;
    if (i < E_)      counts[i] = 0;
}

// ---------------------------------------------------------------------------
// Kernel 1: routing dispatch — build per-expert (token, weight) row lists.
// ---------------------------------------------------------------------------
__global__ __launch_bounds__(256) void moe_dispatch(const int* __restrict__ topk_idx,
                                                    const float* __restrict__ topk_w,
                                                    int* __restrict__ row_token,
                                                    float* __restrict__ row_weight,
                                                    int* __restrict__ counts) {
    const int i = blockIdx.x * blockDim.x + threadIdx.x; // slot id in [0, T*K)
    if (i >= T_ * K_) return;
    const int   e = topk_idx[i];
    const float w = topk_w[i];
    const int   t = i / K_;
    const int pos = atomicAdd(&counts[e], 1);
    if (pos < C_) {
        row_token[e * C_ + pos]  = t;
        row_weight[e * C_ + pos] = w;
    }
}

// ---------------------------------------------------------------------------
// Shared helper: stage a 32(k) x 64(n) f32 weight tile transposed into LDS
// as bf16 [n][k] with row stride LDA. Thread map: k-pair x 4 n-columns so the
// two k-adjacent bf16 pack into one dword (ds_store_b32).
// ---------------------------------------------------------------------------
__device__ __forceinline__ void stage_w_tile(bf16* __restrict__ sB,
                                             const float* __restrict__ w,
                                             int tid, int ldw) {
    const int kp = tid >> 4;           // 0..15 -> k rows 2kp, 2kp+1
    const int ff = (tid & 15) * 4;     // 0..60
    const float* r0 = w + (size_t)(2 * kp) * ldw + ff;
    const float* r1 = r0 + ldw;
    float4 a = *(const float4*)r0;
    float4 b = *(const float4*)r1;
    Pack2 p;
    p.h[0] = (bf16)a.x; p.h[1] = (bf16)b.x; *(unsigned*)&sB[(ff + 0) * LDA + 2 * kp] = p.u;
    p.h[0] = (bf16)a.y; p.h[1] = (bf16)b.y; *(unsigned*)&sB[(ff + 1) * LDA + 2 * kp] = p.u;
    p.h[0] = (bf16)a.z; p.h[1] = (bf16)b.z; *(unsigned*)&sB[(ff + 2) * LDA + 2 * kp] = p.u;
    p.h[0] = (bf16)a.w; p.h[1] = (bf16)b.w; *(unsigned*)&sB[(ff + 3) * LDA + 2 * kp] = p.u;
}

// ---------------------------------------------------------------------------
// Kernel 2: h[e,c,f] = silu(x@Wg) * (x@Wu), bf16 WMMA, f32 accumulate.
// Block = 256 threads (8 waves). Tile: 128 rows (c) x 64 cols (f).
// Each wave: 16 rows x 64 cols = 4 gate + 4 up accumulators.
// ---------------------------------------------------------------------------
__global__ __launch_bounds__(256) void moe_gateup(const float* __restrict__ x,
                                                  const float* __restrict__ wg,
                                                  const float* __restrict__ wu,
                                                  const int* __restrict__ row_token,
                                                  const int* __restrict__ counts,
                                                  bf16* __restrict__ hbuf) {
    const int e  = blockIdx.z;
    const int c0 = blockIdx.y * 128;
    const int f0 = blockIdx.x * 64;
    int cnt = counts[e]; if (cnt > C_) cnt = C_;
    if (c0 >= cnt) return;               // uniform across block

    __shared__ __attribute__((aligned(16))) bf16 sA [128 * LDA]; // [row][k]
    __shared__ __attribute__((aligned(16))) bf16 sBg[64 * LDA];  // transposed [f][k]
    __shared__ __attribute__((aligned(16))) bf16 sBu[64 * LDA];  // transposed [f][k]
    __shared__ int stok[128];

    const int tid = threadIdx.x;
    if (tid < 128) stok[tid] = row_token[e * C_ + c0 + tid];
    __syncthreads();

    const int lane = tid & 31;
    const int wid  = tid >> 5;
    const int half = lane >> 4;
    const int m    = lane & 15;
    const int rowA = wid * 16 + m;

    v8f accg[4] = {}, accu[4] = {};

    for (int k0 = 0; k0 < D_; k0 += 32) {
        // ---- stage A: gather 128 token rows x 32 k, f32 -> bf16, packed ----
        {
            const int r  = tid >> 1;
            const int kk = (tid & 1) * 16;
            const int t  = stok[r];
            Pack16 tmp;
            if (t >= 0) {
                const float* src = x + (size_t)t * D_ + k0 + kk;
                #pragma unroll
                for (int i = 0; i < 16; i += 4) {
                    float4 v = *(const float4*)(src + i);
                    tmp.h[i + 0] = (bf16)v.x; tmp.h[i + 1] = (bf16)v.y;
                    tmp.h[i + 2] = (bf16)v.z; tmp.h[i + 3] = (bf16)v.w;
                }
            } else {
                tmp.q[0] = make_uint4(0u, 0u, 0u, 0u);
                tmp.q[1] = make_uint4(0u, 0u, 0u, 0u);
            }
            *(uint4*)&sA[r * LDA + kk]     = tmp.q[0];
            *(uint4*)&sA[r * LDA + kk + 8] = tmp.q[1];
        }
        // ---- stage B (gate & up): 32 k x 64 f, transposed, packed ----
        stage_w_tile(sBg, wg + ((size_t)e * D_ + k0) * F_ + f0, tid, F_);
        stage_w_tile(sBu, wu + ((size_t)e * D_ + k0) * F_ + f0, tid, F_);
        __syncthreads();

        // ---- fragments + WMMA ----
        Frag16 a, bg, bu;
        a.q[0] = *(const uint4*)&sA[rowA * LDA + 16 * half];      // K = 16h..16h+7
        a.q[1] = *(const uint4*)&sA[rowA * LDA + 16 * half + 8];  // K = 16h+8..
        #pragma unroll
        for (int nt = 0; nt < 4; ++nt) {
            const int ncol = nt * 16 + m;
            bg.q[0] = *(const uint4*)&sBg[ncol * LDA + 16 * half];     // K=16h..16h+7
            bg.q[1] = *(const uint4*)&sBg[ncol * LDA + 16 * half + 8]; // K=16h+8..
            accg[nt] = __builtin_amdgcn_wmma_f32_16x16x32_bf16(
                false, a.v, false, bg.v, (short)0, accg[nt], false, false);
            bu.q[0] = *(const uint4*)&sBu[ncol * LDA + 16 * half];
            bu.q[1] = *(const uint4*)&sBu[ncol * LDA + 16 * half + 8];
            accu[nt] = __builtin_amdgcn_wmma_f32_16x16x32_bf16(
                false, a.v, false, bu.v, (short)0, accu[nt], false, false);
        }
        __syncthreads();
    }

    // ---- epilogue: h = silu(g) * u -> bf16 workspace ----
    #pragma unroll
    for (int nt = 0; nt < 4; ++nt) {
        #pragma unroll
        for (int r = 0; r < 8; ++r) {
            const float g = accg[nt][r];
            const float u = accu[nt][r];
            const float h = (g / (1.0f + __expf(-g))) * u;
            const int c = c0 + wid * 16 + r + 8 * half;   // C/D layout: M = r + 8*half
            const int f = f0 + nt * 16 + m;               // N = lane & 15
            hbuf[((size_t)e * C_ + c) * F_ + f] = (bf16)h;
        }
    }
}

// ---------------------------------------------------------------------------
// Kernel 3: y = h @ Wdown, weighted atomic scatter-add into out[t, :].
// Same tiling: 128 rows (c) x 64 cols (d), K over F in 16 steps of 32.
// A tile is a pure bf16 copy -> async global->LDS path when available.
// ---------------------------------------------------------------------------
__global__ __launch_bounds__(256) void moe_down_combine(const bf16* __restrict__ hbuf,
                                                        const float* __restrict__ wd,
                                                        const int* __restrict__ row_token,
                                                        const float* __restrict__ row_weight,
                                                        const int* __restrict__ counts,
                                                        float* __restrict__ out) {
    const int e  = blockIdx.z;
    const int c0 = blockIdx.y * 128;
    const int d0 = blockIdx.x * 64;
    int cnt = counts[e]; if (cnt > C_) cnt = C_;
    if (c0 >= cnt) return;

    __shared__ __attribute__((aligned(16))) bf16 sA[128 * LDA];  // [row][k]
    __shared__ __attribute__((aligned(16))) bf16 sB[64 * LDA];   // transposed [d][k]
    __shared__ int   stok[128];
    __shared__ float swgt[128];

    const int tid = threadIdx.x;
    if (tid < 128) {
        stok[tid] = row_token[e * C_ + c0 + tid];
        swgt[tid] = row_weight[e * C_ + c0 + tid];
    }
    __syncthreads();

    const int lane = tid & 31;
    const int wid  = tid >> 5;
    const int half = lane >> 4;
    const int m    = lane & 15;
    const int rowA = wid * 16 + m;

    v8f acc[4] = {};

    for (int k0 = 0; k0 < F_; k0 += 32) {
        // ---- stage A: 128 rows x 32 f of h (already bf16), 32B per thread ----
        {
            const int r  = tid >> 1;
            const int kk = (tid & 1) * 16;
            const bf16* src = hbuf + ((size_t)e * C_ + c0 + r) * F_ + k0 + kk;
#ifdef USE_ASYNC_LDS_COPY
            gp_i4 gsrc = (gp_i4)(uintptr_t)src;
            lp_i4 ldst = (lp_i4)(unsigned)(uintptr_t)&sA[r * LDA + kk];
            __builtin_amdgcn_global_load_async_to_lds_b128(gsrc, ldst, 0, 0);
            __builtin_amdgcn_global_load_async_to_lds_b128(gsrc, ldst, 16, 0);
#else
            const uint4* s4 = (const uint4*)src;
            *(uint4*)&sA[r * LDA + kk]     = s4[0];
            *(uint4*)&sA[r * LDA + kk + 8] = s4[1];
#endif
        }
        // ---- stage B: 32 f x 64 d of w_down, transposed f32 -> bf16 ----
        stage_w_tile(sB, wd + ((size_t)e * F_ + k0) * D_ + d0, tid, D_);
#ifdef USE_ASYNC_LDS_COPY
        wait_async0();
#endif
        __syncthreads();

        Frag16 a, b;
        a.q[0] = *(const uint4*)&sA[rowA * LDA + 16 * half];
        a.q[1] = *(const uint4*)&sA[rowA * LDA + 16 * half + 8];
        #pragma unroll
        for (int nt = 0; nt < 4; ++nt) {
            const int ncol = nt * 16 + m;
            b.q[0] = *(const uint4*)&sB[ncol * LDA + 16 * half];
            b.q[1] = *(const uint4*)&sB[ncol * LDA + 16 * half + 8];
            acc[nt] = __builtin_amdgcn_wmma_f32_16x16x32_bf16(
                false, a.v, false, b.v, (short)0, acc[nt], false, false);
        }
        __syncthreads();
    }

    // ---- epilogue: weighted atomic scatter-add (2 commutative adds/elem) ----
    #pragma unroll
    for (int nt = 0; nt < 4; ++nt) {
        #pragma unroll
        for (int r = 0; r < 8; ++r) {
            const int cl = wid * 16 + r + 8 * half;
            const int t  = stok[cl];
            if (t >= 0) {
                const float val = acc[nt][r] * swgt[cl];
                unsafeAtomicAdd(&out[(size_t)t * D_ + d0 + nt * 16 + m], val);
            }
        }
    }
}

// ---------------------------------------------------------------------------
// Host-side launch.
// ---------------------------------------------------------------------------
extern "C" void kernel_launch(void* const* d_in, const int* in_sizes, int n_in,
                              void* d_out, int out_size, void* d_ws, size_t ws_size,
                              hipStream_t stream) {
    const float* x  = (const float*)d_in[0];   // [T, D]
    const int*   ti = (const int*)d_in[1];     // [T, K]
    const float* tw = (const float*)d_in[2];   // [T, K]
    const float* wg = (const float*)d_in[3];   // [E, D, F]
    const float* wu = (const float*)d_in[4];   // [E, D, F]
    const float* wd = (const float*)d_in[5];   // [E, F, D]
    float* out = (float*)d_out;                // [T, D]

    char* ws = (char*)d_ws;
    int*   row_token  = (int*)ws;    ws += (size_t)E_ * C_ * sizeof(int);
    float* row_weight = (float*)ws;  ws += (size_t)E_ * C_ * sizeof(float);
    int*   counts     = (int*)ws;    ws += 256;
    bf16*  hbuf       = (bf16*)ws;   // [E, C, F] bf16 = 33.5 MB

    const int n_init = T_ * D_;
    moe_init<<<(n_init + 255) / 256, 256, 0, stream>>>(out, row_token, counts);
    moe_dispatch<<<(T_ * K_ + 255) / 256, 256, 0, stream>>>(ti, tw, row_token,
                                                            row_weight, counts);
    dim3 gC(F_ / 64, C_ / 128, E_);
    moe_gateup<<<gC, 256, 0, stream>>>(x, wg, wu, row_token, counts, hbuf);
    dim3 gD(D_ / 64, C_ / 128, E_);
    moe_down_combine<<<gD, 256, 0, stream>>>(hbuf, wd, row_token, row_weight,
                                             counts, out);
}
